// PostCrossAttention_89876485636230
// MI455X (gfx1250) — matrix operand order repaired and checked by reading, not compiled, not run on hardware
//
#include <hip/hip_runtime.h>

// Problem constants (from the reference)
#define BATCH 4
#define SEQ   2048
#define HEADS 8
#define CQ    768          // 3*DIM
#define CV    512          // 2*DIM
#define DH    96           // CQ / HEADS
#define DV    64           // CV / HEADS
#define SCALE 0.17677669529663687f   // (DIM/NHEAD)^-0.5 = 32^-0.5
#define AWAVES 4           // waves per attention workgroup

typedef __attribute__((ext_vector_type(16))) __bf16 bf16x16;
typedef __attribute__((ext_vector_type(8)))  __bf16 bf16x8;
typedef __attribute__((ext_vector_type(8)))  float  f32x8;
typedef __attribute__((ext_vector_type(4)))  unsigned int uint32x4;
typedef __attribute__((ext_vector_type(8)))  int int32x8;
typedef __attribute__((ext_vector_type(4)))  int int32x4;

#if defined(__has_builtin)
#if __has_builtin(__builtin_amdgcn_tensor_load_to_lds) && __has_builtin(__builtin_amdgcn_s_wait_tensorcnt)
#define HAS_TDM 1
#endif
#endif
#ifndef HAS_TDM
#define HAS_TDM 0
#endif

// ---- WMMA fragment loaders (contiguous k = 16h+s order, wave32) -------------
// A (16x32, 16-bit): lane L holds row M=L%16; lane-half h=L/16 selects
// K in {16h..16h+15}  -> one contiguous 32B load.
static __device__ inline bf16x16 load_a_frag(const __bf16* base, int rowStride,
                                             int row, int k0, int half) {
  const __bf16* p = base + (size_t)row * rowStride + k0 + 16 * half;
  return *(const bf16x16*)p;
}
// B (32x16, 16-bit): lane L holds column N=L%16; lane-half selects
// K in {0..15} or {16..31} -> one contiguous 32B load of Bt[n][k] (n-major).
static __device__ inline bf16x16 load_b_frag(const __bf16* base, int rowStride,
                                             int row, int k0, int half) {
  const __bf16* p = base + (size_t)row * rowStride + k0 + 16 * half;
  return *(const bf16x16*)p;
}
static __device__ inline f32x8 wmma_bf16(bf16x16 a, bf16x16 b, f32x8 c) {
  return __builtin_amdgcn_wmma_f32_16x16x32_bf16(false, a, false, b,
                                                 (short)0, c, false, false);
}

#if HAS_TDM
// ---- TDM: async 2D tile load (bf16 elements) global -> LDS ---------------
// Builds D# group0/group1 per CDNA5 ISA §8.3/§8.4; groups 2/3 zero (2D only).
// Toolchain here is the 6-arg builtin: (g0, g1, g2, g3, g4, cpol).
static __device__ inline void tdm_load_2d_bf16(unsigned lds_off, const __bf16* gptr,
                                               unsigned tile_w, unsigned tile_h,
                                               unsigned stride_elems) {
  unsigned long long ga = (unsigned long long)(size_t)gptr;
  uint32x4 g0;
  g0[0] = 1u;                                   // count=1 (valid), user mode
  g0[1] = lds_off;                              // lds_addr  (bits 63:32)
  g0[2] = (unsigned)(ga & 0xffffffffu);         // global_addr[31:0]
  g0[3] = (unsigned)((ga >> 32) & 0x01ffffffu)  // global_addr[56:32]
        | (2u << 30);                           // type = 2 ("image")
  int32x8 g1;
  g1[0] = (int)(1u << 16);                      // wg_mask=0, data_size=1 (2B)
  g1[1] = (int)((tile_w & 0xffffu) << 16);      // tensor_dim0[15:0]
  g1[2] = (int)((tile_w >> 16) | ((tile_h & 0xffffu) << 16)); // dim0 hi | dim1 lo
  g1[3] = (int)((tile_h >> 16) | (tile_w << 16));             // dim1 hi | tile_dim0
  g1[4] = (int)(tile_h & 0xffffu);              // tile_dim1 | tile_dim2=0
  g1[5] = (int)stride_elems;                    // tensor_dim0_stride[31:0]
  g1[6] = 0;                                    // stride0 hi | stride1 lo
  g1[7] = 0;                                    // stride1 hi
  int32x4 z4 = {0, 0, 0, 0};
  int32x8 z8 = {0, 0, 0, 0, 0, 0, 0, 0};
  __builtin_amdgcn_tensor_load_to_lds(g0, g1, z4, z4, z8, 0);
}
#endif

// ---- f32 -> bf16 conversion ----------------------------------------------
__global__ void cvt_f32_bf16(const float* __restrict src, __bf16* __restrict dst, int n) {
  int i = blockIdx.x * blockDim.x + threadIdx.x;
  if (i < n) dst[i] = (__bf16)src[i];
}

// ---- Projection GEMM: out = X @ W^T  (X: MxK, W: NxK, both row-major bf16) -
// One wave computes a 16x64 tile. mode 0: scatter to (B,H,SEQ,headDim);
// mode 1: scatter transposed to (B,H,headDim,SEQ).
__global__ __launch_bounds__(32)
void proj_gemm(const __bf16* __restrict X, const __bf16* __restrict W,
               __bf16* __restrict out, int M, int N, int K, int mode, int headDim) {
  const int lane = threadIdx.x & 31;
  const int half = lane >> 4;
  const int l16  = lane & 15;
  const int tilesN = N >> 6;
  const int tm = (blockIdx.x / tilesN) * 16;
  const int tn = (blockIdx.x % tilesN) * 64;

  f32x8 acc[4] = {};
  for (int k0 = 0; k0 < K; k0 += 32) {
    bf16x16 a = load_a_frag(X, K, tm + l16, k0, half);
#pragma unroll
    for (int t = 0; t < 4; ++t) {
      bf16x16 b = load_b_frag(W, K, tn + t * 16 + l16, k0, half);
      acc[t] = wmma_bf16(a, b, acc[t]);
    }
  }
#pragma unroll
  for (int t = 0; t < 4; ++t) {
#pragma unroll
    for (int r = 0; r < 8; ++r) {
      int m = tm + r + 8 * half;
      int n = tn + t * 16 + l16;
      int b    = m / SEQ, nrow = m % SEQ;
      int hh   = n / headDim, d = n % headDim;
      size_t idx;
      if (mode == 0) idx = (((size_t)(b * HEADS + hh)) * SEQ + nrow) * headDim + d;
      else           idx = (((size_t)(b * HEADS + hh)) * headDim + d) * SEQ + nrow;
      out[idx] = (__bf16)acc[t][r];
    }
  }
}

// ---- Fused flash attention with post-softmax mask renormalization ---------
// 4 waves / workgroup handle 64 query rows of one (b,h). Per 32-key block the
// K tile (32x96) and V^T tile (64x32) are staged into LDS once (TDM async
// tensor load when available), then all 4 waves consume them.
__global__ __launch_bounds__(AWAVES * 32)
void flash_attn(const __bf16* __restrict Q, const __bf16* __restrict Kp,
                const __bf16* __restrict Vt, const float* __restrict masks,
                float* __restrict out) {
  __shared__ alignas(64) __bf16 Klds[32 * DH];            // 6 KB  [key][dh]
  __shared__ alignas(64) __bf16 Vlds[DV * 32];            // 4 KB  [dv][key]
  __shared__ alignas(32) __bf16 Plds[AWAVES][16 * 32];    // 4 KB  per-wave P
  const int wave = threadIdx.x >> 5;
  const int lane = threadIdx.x & 31;
  const int half = lane >> 4;
  const int l16  = lane & 15;

  const int rowGroups = SEQ / (16 * AWAVES);
  const int rg = blockIdx.x % rowGroups;
  const int bh = blockIdx.x / rowGroups;
  const int b  = bh / HEADS;
  const int hh = bh % HEADS;
  const int r0 = rg * 16 * AWAVES + wave * 16;

  const __bf16* Qb = Q  + (size_t)bh * SEQ * DH;
  const __bf16* Kb = Kp + (size_t)bh * SEQ * DH;
  const __bf16* Vb = Vt + (size_t)bh * DV * SEQ;
  const float*  Mb = masks + (size_t)b * SEQ * SEQ;

#if HAS_TDM
  const unsigned klds_off = (unsigned)(size_t)(void*)&Klds[0];
  const unsigned vlds_off = (unsigned)(size_t)(void*)&Vlds[0];
#endif

  // Q fragments for this wave's row tile (dh = 96 = 3 K-steps of 32)
  bf16x16 qa[3];
#pragma unroll
  for (int kk = 0; kk < 3; ++kk) qa[kk] = load_a_frag(Qb, DH, r0 + l16, kk * 32, half);

  f32x8 acc[4] = {};                       // 16 x 64 output accumulator
  float rm[8], lsum[8], msum[8];           // per-row (r -> row r0+r+8*half)
#pragma unroll
  for (int r = 0; r < 8; ++r) { rm[r] = -1e30f; lsum[r] = 0.f; msum[r] = 0.f; }

  for (int j = 0; j < SEQ; j += 32) {
    // ---- stage K tile (32 keys x 96) and V^T tile (64 dv x 32 keys) -> LDS
#if HAS_TDM
    if (wave == 0) {
      tdm_load_2d_bf16(klds_off, Kb + (size_t)j * DH, DH, 32, DH);
      tdm_load_2d_bf16(vlds_off, Vb + j, 32, DV, SEQ);
      __builtin_amdgcn_s_wait_tensorcnt(0);
    }
#else
    // cooperative synchronous copy fallback
    {
      const uint4* ks = (const uint4*)(Kb + (size_t)j * DH);   // 32 rows contiguous
      uint4* kd = (uint4*)&Klds[0];
      for (int i = threadIdx.x; i < (32 * DH) / 8; i += AWAVES * 32) kd[i] = ks[i];
      for (int i = threadIdx.x; i < DV * 4; i += AWAVES * 32) {
        int row = i >> 2, c = i & 3;
        ((uint4*)(&Vlds[row * 32]))[c] = ((const uint4*)(Vb + (size_t)row * SEQ + j))[c];
      }
    }
#endif
    __syncthreads();
    if (j + 32 < SEQ) __builtin_prefetch(Mb + (size_t)r0 * SEQ + j + 32, 0, 1);

    // ---- S = Q K^T tile (16 x 32), two 16-col subtiles, K frags from LDS
    f32x8 s[2] = {};
#pragma unroll
    for (int t = 0; t < 2; ++t)
#pragma unroll
      for (int kk = 0; kk < 3; ++kk) {
        bf16x16 kb = load_b_frag(&Klds[0], DH, t * 16 + l16, kk * 32, half);
        s[t] = wmma_bf16(qa[kk], kb, s[t]);
      }

    // mask values at C-layout positions
    float mk[2][8], p[2][8];
#pragma unroll
    for (int t = 0; t < 2; ++t)
#pragma unroll
      for (int r = 0; r < 8; ++r)
        mk[t][r] = Mb[(size_t)(r0 + r + 8 * half) * SEQ + (j + t * 16 + l16)];

    // block row-max (reduce across the 16 lanes of each half-wave)
    float bm[8];
#pragma unroll
    for (int r = 0; r < 8; ++r) bm[r] = fmaxf(s[0][r], s[1][r]) * SCALE;
#pragma unroll
    for (int d = 1; d < 16; d <<= 1)
#pragma unroll
      for (int r = 0; r < 8; ++r) bm[r] = fmaxf(bm[r], __shfl_xor(bm[r], d, 32));

    float alpha[8];
#pragma unroll
    for (int r = 0; r < 8; ++r) {
      float nm = fmaxf(rm[r], bm[r]);
      alpha[r] = __expf(rm[r] - nm);
      rm[r] = nm;
    }
#pragma unroll
    for (int t = 0; t < 2; ++t)
#pragma unroll
      for (int r = 0; r < 8; ++r) p[t][r] = __expf(s[t][r] * SCALE - rm[r]);

    // row sums: Z (plain exp sum) and mask sum
    float psum[8], msadd[8];
#pragma unroll
    for (int r = 0; r < 8; ++r) { psum[r] = p[0][r] + p[1][r]; msadd[r] = mk[0][r] + mk[1][r]; }
#pragma unroll
    for (int d = 1; d < 16; d <<= 1)
#pragma unroll
      for (int r = 0; r < 8; ++r) {
        psum[r]  += __shfl_xor(psum[r],  d, 32);
        msadd[r] += __shfl_xor(msadd[r], d, 32);
      }
#pragma unroll
    for (int r = 0; r < 8; ++r) {
      lsum[r] = lsum[r] * alpha[r] + psum[r];
      msum[r] += msadd[r];
    }
#pragma unroll
    for (int t = 0; t < 4; ++t)
#pragma unroll
      for (int r = 0; r < 8; ++r) acc[t][r] *= alpha[r];

    // P*mask -> bf16 -> per-wave LDS slice (C-layout scatter -> A layout read)
#pragma unroll
    for (int t = 0; t < 2; ++t)
#pragma unroll
      for (int r = 0; r < 8; ++r)
        Plds[wave][(r + 8 * half) * 32 + t * 16 + l16] = (__bf16)(p[t][r] * mk[t][r]);

    bf16x16 pa = load_a_frag(&Plds[wave][0], 32, l16, 0, half);
#pragma unroll
    for (int t = 0; t < 4; ++t) {
      bf16x16 vb = load_b_frag(&Vlds[0], 32, t * 16 + l16, 0, half);
      acc[t] = wmma_bf16(pa, vb, acc[t]);
    }
    __syncthreads();   // all waves done with Klds/Vlds before next stage
  }

  float inv[8];
#pragma unroll
  for (int r = 0; r < 8; ++r) inv[r] = 1.0f / (lsum[r] * (float)HEADS * msum[r] + 1e-30f);
#pragma unroll
  for (int t = 0; t < 4; ++t)
#pragma unroll
    for (int r = 0; r < 8; ++r) {
      int nrow = r0 + r + 8 * half;
      out[((size_t)b * SEQ + nrow) * CV + hh * DV + t * 16 + l16] = acc[t][r] * inv[r];
    }
}

// ---- Host launcher --------------------------------------------------------
extern "C" void kernel_launch(void* const* d_in, const int* in_sizes, int n_in,
                              void* d_out, int out_size, void* d_ws, size_t ws_size,
                              hipStream_t stream) {
  (void)in_sizes; (void)n_in; (void)out_size; (void)ws_size;
  const float* q     = (const float*)d_in[0];
  const float* k     = (const float*)d_in[1];
  const float* v     = (const float*)d_in[2];
  const float* masks = (const float*)d_in[3];
  const float* Wq    = (const float*)d_in[4];
  const float* Wk    = (const float*)d_in[5];
  const float* Wv    = (const float*)d_in[6];
  float* out = (float*)d_out;

  // Workspace carve (~70 MB total, 256B aligned slices)
  char*  ws  = (char*)d_ws;
  size_t off = 0;
  auto carve = [&](size_t elems) {
    char* p = ws + off;
    off = (off + elems * sizeof(__bf16) + 255) & ~(size_t)255;
    return (__bf16*)p;
  };
  const int nQ  = BATCH * SEQ * CQ;       // 6.29M
  const int nV  = BATCH * SEQ * CV;       // 4.19M
  const int nWq = CQ * CQ, nWv = CV * CV;
  __bf16* qbf  = carve(nQ);
  __bf16* kbf  = carve(nQ);
  __bf16* vbf  = carve(nV);
  __bf16* Wqb  = carve(nWq);
  __bf16* Wkb  = carve(nWq);
  __bf16* Wvb  = carve(nWv);
  __bf16* Qp   = carve(nQ);               // (B,H,SEQ,96)
  __bf16* Kp   = carve(nQ);               // (B,H,SEQ,96)
  __bf16* Vtp  = carve(nV);               // (B,H,64,SEQ)

  cvt_f32_bf16<<<(nQ  + 255) / 256, 256, 0, stream>>>(q,  qbf, nQ);
  cvt_f32_bf16<<<(nQ  + 255) / 256, 256, 0, stream>>>(k,  kbf, nQ);
  cvt_f32_bf16<<<(nV  + 255) / 256, 256, 0, stream>>>(v,  vbf, nV);
  cvt_f32_bf16<<<(nWq + 255) / 256, 256, 0, stream>>>(Wq, Wqb, nWq);
  cvt_f32_bf16<<<(nWq + 255) / 256, 256, 0, stream>>>(Wk, Wkb, nWq);
  cvt_f32_bf16<<<(nWv + 255) / 256, 256, 0, stream>>>(Wv, Wvb, nWv);

  const int M = BATCH * SEQ;              // 8192
  proj_gemm<<<(M / 16) * (CQ / 64), 32, 0, stream>>>(qbf, Wqb, Qp,  M, CQ, CQ, 0, DH);
  proj_gemm<<<(M / 16) * (CQ / 64), 32, 0, stream>>>(kbf, Wkb, Kp,  M, CQ, CQ, 0, DH);
  proj_gemm<<<(M / 16) * (CV / 64), 32, 0, stream>>>(vbf, Wvb, Vtp, M, CV, CV, 1, DV);

  flash_attn<<<BATCH * HEADS * (SEQ / (16 * AWAVES)), AWAVES * 32, 0, stream>>>(
      Qp, Kp, Vtp, masks, out);
}
